// FrequencyBranch_6932077215921
// MI455X (gfx1250) — matrix operand order repaired and checked by reading, not compiled, not run on hardware
//
#include <hip/hip_runtime.h>
#include <math.h>

typedef __bf16 bf16;
typedef __attribute__((ext_vector_type(16))) __bf16 v16bf;
typedef __attribute__((ext_vector_type(8)))  __bf16 bf16x8;
typedef __attribute__((ext_vector_type(8)))  float  v8f;

#define LDS_STRIDE 129   // 128 + 1 pad: row pass is bank-conflict-free

// ---------------------------------------------------------------------------
// Kernel 1: per (batch,channel) 128x128 2D FFT -> fftshift -> amp/phase -> bf16
// ---------------------------------------------------------------------------
__global__ void __launch_bounds__(128) fft_feats_kernel(const float* __restrict__ img,
                                                        bf16* __restrict__ feats)
{
    extern __shared__ float smem[];
    float* re = smem;
    float* im = smem + 128 * LDS_STRIDE;
    const int  tid = threadIdx.x;                 // 0..127
    const long bc  = blockIdx.x;                  // b*3 + c
    const float* src = img + bc * 16384;

    // Coalesced load; columns bit-reversed so the row DIT FFT outputs natural order.
    for (int idx = tid; idx < 16384; idx += 128) {
        int u = idx >> 7, v = idx & 127;
        int vr = (int)(__brev((unsigned)v) >> 25);   // 7-bit reverse
        re[u * LDS_STRIDE + vr] = src[idx];
        im[u * LDS_STRIDE + vr] = 0.0f;
    }
    __syncthreads();

    // ---- row FFTs: thread t owns row t ----
    {
        float* rr = re + tid * LDS_STRIDE;
        float* ri = im + tid * LDS_STRIDE;
        for (int len = 2; len <= 128; len <<= 1) {
            const int   half = len >> 1;
            const float ang  = -6.28318530717958647692f / (float)len;
            for (int i = 0; i < 128; i += len) {
                for (int j = 0; j < half; ++j) {
                    float s, c;
                    __sincosf(ang * (float)j, &s, &c);
                    const int a = i + j, b = a + half;
                    float tr = c * rr[b] - s * ri[b];
                    float ti = c * ri[b] + s * rr[b];
                    rr[b] = rr[a] - tr;  ri[b] = ri[a] - ti;
                    rr[a] += tr;         ri[a] += ti;
                }
            }
        }
    }
    __syncthreads();

    // ---- column FFTs: thread t owns column t ----
    {
        const int t = tid;
        for (int i = 0; i < 128; ++i) {           // bit-reverse along rows
            int j = (int)(__brev((unsigned)i) >> 25);
            if (j > i) {
                float x;
                x = re[i*LDS_STRIDE+t]; re[i*LDS_STRIDE+t] = re[j*LDS_STRIDE+t]; re[j*LDS_STRIDE+t] = x;
                x = im[i*LDS_STRIDE+t]; im[i*LDS_STRIDE+t] = im[j*LDS_STRIDE+t]; im[j*LDS_STRIDE+t] = x;
            }
        }
        for (int len = 2; len <= 128; len <<= 1) {
            const int   half = len >> 1;
            const float ang  = -6.28318530717958647692f / (float)len;
            for (int i = 0; i < 128; i += len) {
                for (int j = 0; j < half; ++j) {
                    float s, c;
                    __sincosf(ang * (float)j, &s, &c);
                    const int a = (i + j) * LDS_STRIDE + t;
                    const int b = a + half * LDS_STRIDE;
                    float tr = c * re[b] - s * im[b];
                    float ti = c * im[b] + s * re[b];
                    re[b] = re[a] - tr;  im[b] = im[a] - ti;
                    re[a] += tr;         im[a] += ti;
                }
            }
        }
    }
    __syncthreads();

    // ---- fftshift + amp/phase, bf16 store ----
    bf16* amp = feats + bc * 32768;
    bf16* ph  = amp + 16384;
    for (int idx = tid; idx < 16384; idx += 128) {
        int u = idx >> 7, v = idx & 127;
        int us = (u + 64) & 127, vs = (v + 64) & 127;
        float r  = re[us * LDS_STRIDE + vs];
        float ii = im[us * LDS_STRIDE + vs];
        amp[idx] = (bf16)sqrtf(r * r + ii * ii);
        ph[idx]  = (bf16)atan2f(ii, r);
    }
}

// ---------------------------------------------------------------------------
// WMMA fragment plumbing (contiguous k order k = 16*g + s, same in A and B).
// B: lane covers K = kb + g*16 + [0..15] of W column n (16 contiguous fp32 -> bf16)
// A: row m, K = kb + g*16 + [0..15] (two 16 B loads)
// ---------------------------------------------------------------------------
template<int TM>
struct KTile { v16bf b; v16bf a[TM]; };

template<int TM>
__device__ __forceinline__ KTile<TM> load_ktile(const bf16* __restrict__ A,
                                                const float* __restrict__ wp,
                                                int kb, size_t arow0, int K, int g)
{
    KTile<TM> t;
    const float4* w4 = (const float4*)(wp + kb);
    float4 f0 = w4[0], f1 = w4[1], f2 = w4[2], f3 = w4[3];
    union { v16bf v; bf16 e[16]; } bu;
    bu.e[ 0]=(bf16)f0.x; bu.e[ 1]=(bf16)f0.y; bu.e[ 2]=(bf16)f0.z; bu.e[ 3]=(bf16)f0.w;
    bu.e[ 4]=(bf16)f1.x; bu.e[ 5]=(bf16)f1.y; bu.e[ 6]=(bf16)f1.z; bu.e[ 7]=(bf16)f1.w;
    bu.e[ 8]=(bf16)f2.x; bu.e[ 9]=(bf16)f2.y; bu.e[10]=(bf16)f2.z; bu.e[11]=(bf16)f2.w;
    bu.e[12]=(bf16)f3.x; bu.e[13]=(bf16)f3.y; bu.e[14]=(bf16)f3.z; bu.e[15]=(bf16)f3.w;
    t.b = bu.v;
#pragma unroll
    for (int i = 0; i < TM; ++i) {
        const bf16* ap = A + arow0 + (size_t)i * 16 * K + kb + g * 16;
        union { v16bf v; bf16x8 h[2]; } au;
        au.h[0] = *(const bf16x8*)(ap);
        au.h[1] = *(const bf16x8*)(ap + 8);
        t.a[i] = au.v;
    }
    return t;
}

template<int TM>
__device__ __forceinline__ void wmma_all(const KTile<TM>& t, v8f acc[TM])
{
#pragma unroll
    for (int i = 0; i < TM; ++i)
        acc[i] = __builtin_amdgcn_wmma_f32_16x16x32_bf16(
                     false, t.a[i], false, t.b, (short)0, acc[i], false, false);
}

// Ping-pong double-buffered k-loop: each buffer is refilled in place right after
// it is consumed, so no cross-buffer register copies (requires KC/32 even, >= 2).
template<int TM>
__device__ __forceinline__ void gemm_core(const bf16* __restrict__ A,
                                          const float* __restrict__ wp,
                                          size_t arow0, int K, int g, int KC,
                                          v8f acc[TM])
{
    KTile<TM> ta = load_ktile<TM>(A, wp, 0,  arow0, K, g);
    KTile<TM> tb = load_ktile<TM>(A, wp, 32, arow0, K, g);
    for (int kb = 64; kb + 32 < KC; kb += 64) {
        wmma_all<TM>(ta, acc);
        ta = load_ktile<TM>(A, wp, kb, arow0, K, g);
        wmma_all<TM>(tb, acc);
        tb = load_ktile<TM>(A, wp, kb + 32, arow0, K, g);
    }
    wmma_all<TM>(ta, acc);
    wmma_all<TM>(tb, acc);
}

// ---------------------------------------------------------------------------
// Kernel 2a: split-K GEMM for layer 1.  parts[kz][M][N] fp32 partial sums.
// Grid: (N/64, M/(TM*16), SK); block (32,4).
// ---------------------------------------------------------------------------
template<int TM>
__global__ void __launch_bounds__(128) gemm_splitk(const bf16* __restrict__ A,
                                                   const float* __restrict__ W,
                                                   float* __restrict__ parts,
                                                   int M, int N, int K, int KC)
{
    const int lane  = threadIdx.x;
    const int g     = lane >> 4;
    const int l16   = lane & 15;
    const int ntile = blockIdx.x * blockDim.y + threadIdx.y;
    const int n     = ntile * 16 + l16;
    const int m0    = blockIdx.y * (TM * 16);
    const int k0    = blockIdx.z * KC;

    const v8f vzero = {0.f, 0.f, 0.f, 0.f, 0.f, 0.f, 0.f, 0.f};
    v8f acc[TM];
#pragma unroll
    for (int i = 0; i < TM; ++i) acc[i] = vzero;

    const float* wp    = W + (size_t)n * K + k0 + g * 16;
    const size_t arow0 = (size_t)(m0 + l16) * K + k0;

    gemm_core<TM>(A, wp, arow0, K, g, KC, acc);

    float* out = parts + (size_t)blockIdx.z * M * N;
#pragma unroll
    for (int i = 0; i < TM; ++i)
#pragma unroll
        for (int r = 0; r < 8; ++r) {
            const int m = m0 + i * 16 + g * 8 + r;
            out[(size_t)m * N + n] = acc[i][r];
        }
}

// Reduce split-K partials + bias + ReLU -> bf16 activations
__global__ void __launch_bounds__(256) reduce_bias_relu(const float* __restrict__ parts,
                                                        const float* __restrict__ bias,
                                                        bf16* __restrict__ X,
                                                        int MN, int N, int SK)
{
    int i = blockIdx.x * blockDim.x + threadIdx.x;
    if (i >= MN) return;
    float s = 0.f;
    for (int k = 0; k < SK; ++k) s += parts[(size_t)k * MN + i];
    s += bias[i % N];
    X[i] = (bf16)fmaxf(s, 0.f);
}

// ---------------------------------------------------------------------------
// Kernel 2b: fused GEMM + bias + activation for the small layers (2 and 3).
// ---------------------------------------------------------------------------
template<int TM, bool RELU, bool OUT_BF16>
__global__ void __launch_bounds__(128) gemm_bias_act(const bf16* __restrict__ A,
                                                     const float* __restrict__ W,
                                                     const float* __restrict__ bias,
                                                     void* __restrict__ Cout,
                                                     int M, int N, int K)
{
    const int lane  = threadIdx.x;
    const int g     = lane >> 4;
    const int l16   = lane & 15;
    const int ntile = blockIdx.x * blockDim.y + threadIdx.y;
    const int n     = ntile * 16 + l16;
    const int m0    = blockIdx.y * (TM * 16);

    const v8f vzero = {0.f, 0.f, 0.f, 0.f, 0.f, 0.f, 0.f, 0.f};
    v8f acc[TM];
#pragma unroll
    for (int i = 0; i < TM; ++i) acc[i] = vzero;

    const float* wp    = W + (size_t)n * K + g * 16;
    const size_t arow0 = (size_t)(m0 + l16) * K;

    gemm_core<TM>(A, wp, arow0, K, g, K, acc);

    const float bv = bias[n];
#pragma unroll
    for (int i = 0; i < TM; ++i)
#pragma unroll
        for (int r = 0; r < 8; ++r) {
            const int m = m0 + i * 16 + g * 8 + r;
            float val = acc[i][r] + bv;
            if (RELU) val = fmaxf(val, 0.0f);
            if (OUT_BF16) ((bf16*)Cout)[(size_t)m * N + n] = (bf16)val;
            else          ((float*)Cout)[(size_t)m * N + n] = val;
        }
}

// ---------------------------------------------------------------------------
extern "C" void kernel_launch(void* const* d_in, const int* in_sizes, int n_in,
                              void* d_out, int out_size, void* d_ws, size_t ws_size,
                              hipStream_t stream)
{
    const float* img = (const float*)d_in[0];   // [256,3,128,128]
    const float* W1  = (const float*)d_in[1];   // [512,98304]
    const float* b1  = (const float*)d_in[2];
    const float* W2  = (const float*)d_in[3];   // [256,512]
    const float* b2  = (const float*)d_in[4];
    const float* W3  = (const float*)d_in[5];   // [128,256]
    const float* b3  = (const float*)d_in[6];
    float* out = (float*)d_out;                 // [256,128] fp32

    constexpr int M = 256, N1 = 512, K1 = 98304, N2 = 256, N3 = 128;
    constexpr int SK = 8, KC = K1 / SK;         // 12288 per split (384 k-steps)

    char* ws = (char*)d_ws;
    bf16*  feats = (bf16*)ws;                               // 256*98304 bf16 = 50,331,648 B
    float* parts = (float*)(ws + 50331648);                 // 8*256*512 fp32 = 4,194,304 B
    bf16*  X1    = (bf16*)(ws + 50331648 + 4194304);        // 256*512 bf16
    bf16*  X2    = (bf16*)(ws + 50331648 + 4194304 + 262144);

    const size_t lds_bytes = (size_t)2 * 128 * LDS_STRIDE * sizeof(float);   // 132,096 B
    (void)hipFuncSetAttribute(reinterpret_cast<const void*>(fft_feats_kernel),
                              hipFuncAttributeMaxDynamicSharedMemorySize,
                              (int)lds_bytes);
    fft_feats_kernel<<<dim3(256 * 3), dim3(128), lds_bytes, stream>>>(img, feats);

    dim3 blk(32, 4);  // 4 waves/block, each wave owns a 16-wide N stripe

    // L1: split-K (256 blocks / 1024 waves -> enough loads in flight for 23.3 TB/s)
    gemm_splitk<4><<<dim3(N1 / 64, M / 64, SK), blk, 0, stream>>>(
        feats, W1, parts, M, N1, K1, KC);
    reduce_bias_relu<<<dim3((M * N1 + 255) / 256), dim3(256), 0, stream>>>(
        parts, b1, X1, M * N1, N1, SK);

    // L2: [256,512] x [512,256] -> relu -> bf16
    gemm_bias_act<4, true,  true ><<<dim3(N2 / 64, M / 64), blk, 0, stream>>>(
        X1, W2, b2, (void*)X2, M, N2, 512);
    // L3: [256,256] x [256,128] + bias -> fp32 out
    gemm_bias_act<4, false, false><<<dim3(N3 / 64, M / 64), blk, 0, stream>>>(
        X2, W3, b3, (void*)out, M, N3, 256);
}
